// NewRoIHeads_37641093382199
// MI455X (gfx1250) — compile-verified, matrix operand order checked by reading, not run on hardware
//
#include <hip/hip_runtime.h>
#include <hip/hip_bf16.h>
#include <math.h>

#define NB    32
#define NP    8000
#define NG    100
#define NPT   8100          // P + G
#define PTPAD 8192          // padded label row stride (pad = -1)
#define KTOT  512
#define NPOS_CAP 128
#define HHALF ((NB * NPT) / 2)   // 129600, threefry counter split point

typedef __attribute__((ext_vector_type(16))) _Float16 v16h;
typedef __attribute__((ext_vector_type(8)))  float    v8f;

// ---------------- CDNA5 async global->LDS helpers (inline asm) -------------
__device__ __forceinline__ unsigned lds_offset(const void* p) {
    // generic -> addrspace(3) cast yields the 32-bit LDS byte offset
    return (unsigned)(size_t)(__attribute__((address_space(3))) const void*)p;
}
__device__ __forceinline__ void async_load_b128(unsigned lds_off, const void* gptr) {
    asm volatile("global_load_async_to_lds_b128 %0, %1, off"
                 :: "v"(lds_off), "v"((unsigned long long)(size_t)gptr)
                 : "memory");
}
__device__ __forceinline__ void wait_asynccnt0() {
    asm volatile("s_wait_asynccnt 0" ::: "memory");
}

// ---------------- JAX threefry2x32, key = (0, 42) -------------------------
__device__ __forceinline__ unsigned int rotl32(unsigned int v, int n) {
    return (v << n) | (v >> (32 - n));
}

__device__ __forceinline__ void threefry2x32(unsigned int k0, unsigned int k1,
                                             unsigned int x0, unsigned int x1,
                                             unsigned int& o0, unsigned int& o1) {
    const unsigned int k2 = k0 ^ k1 ^ 0x1BD11BDAu;
    unsigned int y0 = x0 + k0, y1 = x1 + k1;
    const int RA[4] = {13, 15, 26, 6};
    const int RB[4] = {17, 29, 16, 24};
#pragma unroll
    for (int i = 0; i < 4; ++i) { y0 += y1; y1 = rotl32(y1, RA[i]); y1 ^= y0; }
    y0 += k1; y1 += k2 + 1u;
#pragma unroll
    for (int i = 0; i < 4; ++i) { y0 += y1; y1 = rotl32(y1, RB[i]); y1 ^= y0; }
    y0 += k2; y1 += k0 + 2u;
#pragma unroll
    for (int i = 0; i < 4; ++i) { y0 += y1; y1 = rotl32(y1, RA[i]); y1 ^= y0; }
    y0 += k0; y1 += k1 + 3u;
#pragma unroll
    for (int i = 0; i < 4; ++i) { y0 += y1; y1 = rotl32(y1, RB[i]); y1 ^= y0; }
    y0 += k1; y1 += k2 + 4u;
    o0 = y0; o1 = y1;
}

// ---------------- Kernel 1: match + label + threefry uniform ---------------
__global__ void __launch_bounds__(256)
match_kernel(const float* __restrict__ proposals,
             const float* __restrict__ gt_boxes,
             const int*   __restrict__ gt_labels,
             float* __restrict__ r_out,
             int*   __restrict__ lab_out,
             int*   __restrict__ midx_out) {
    __shared__ float4 sgt[NG] __attribute__((aligned(16)));
    __shared__ float  sarea[NG];
    __shared__ int    slab[NG];
    const int b = blockIdx.y;
    const int t = blockIdx.x * 256 + threadIdx.x;   // t in [0, 8192)

    // async-stage the 100 GT boxes (1.6 KB) into LDS, reused 8100x per row
    if (threadIdx.x < NG) {
        async_load_b128(lds_offset(&sgt[threadIdx.x]),
                        &((const float4*)gt_boxes)[b * NG + threadIdx.x]);
    }
    wait_asynccnt0();
    __syncthreads();
    for (int g = threadIdx.x; g < NG; g += 256) {
        float4 gb = sgt[g];
        sarea[g] = (gb.z - gb.x) * (gb.w - gb.y);
        slab[g]  = gt_labels[b * NG + g];
    }
    __syncthreads();

    int labv = -1;                                  // pad value for t in [NPT, 8192)
    if (t < NPT) {
        float4 pb = (t < NP) ? ((const float4*)proposals)[b * NP + t] : sgt[t - NP];
        const float area_p = (pb.z - pb.x) * (pb.w - pb.y);

        float best = -1.0f; int bidx = 0;
#pragma unroll 4
        for (int g = 0; g < NG; ++g) {
            float4 gb = sgt[g];
            float w = fminf(gb.z, pb.z) - fmaxf(gb.x, pb.x);
            float h = fminf(gb.w, pb.w) - fmaxf(gb.y, pb.y);
            float inter = fmaxf(w, 0.0f) * fmaxf(h, 0.0f);
            float iou = inter / (sarea[g] + area_p - inter);
            if (iou > best) { best = iou; bidx = g; }   // first-max == jnp.argmax
        }
        const int matched = (best < 0.5f) ? -1 : bidx;
        labv              = (matched < 0) ? 0 : slab[bidx];
        const int clamped = (matched < 0) ? 0 : bidx;    // jnp.clip(matched, 0)

        // JAX uniform(key(42), [NB, NPT]) : counters split in halves
        const int e = b * NPT + t;
        const unsigned int i = (e < (int)HHALF) ? (unsigned int)e
                                                : (unsigned int)(e - HHALF);
        unsigned int o0, o1;
        threefry2x32(0u, 42u, i, HHALF + i, o0, o1);
        const unsigned int bits = (e < (int)HHALF) ? o0 : o1;
        r_out[e]    = __uint_as_float((bits >> 9) | 0x3f800000u) - 1.0f;
        midx_out[e] = clamped;
    }
    lab_out[b * PTPAD + t] = labv;   // grid covers exactly [0, PTPAD)
}

// ---------------- Kernel 2: WMMA pos/neg counts (one wave32 per row) -------
__global__ void __launch_bounds__(32)
count_kernel(const int* __restrict__ lab, int* __restrict__ stats) {
    const int b    = blockIdx.x;
    const int lane = threadIdx.x;
    const int4* lrow = (const int4*)(lab + (size_t)b * PTPAD);

    v16h ones;
#pragma unroll
    for (int j = 0; j < 16; ++j) ones[j] = (_Float16)1.0f;

    v8f accp = {}; v8f accn = {};
#pragma unroll 1
    for (int step = 0; step < 16; ++step) {          // 16 * 512 = 8192 == PTPAD
        const int qb = step * 128 + lane * 4;
        const int4 q0 = lrow[qb + 0];
        const int4 q1 = lrow[qb + 1];
        const int4 q2 = lrow[qb + 2];
        const int4 q3 = lrow[qb + 3];
        const int vals[16] = {q0.x, q0.y, q0.z, q0.w, q1.x, q1.y, q1.z, q1.w,
                              q2.x, q2.y, q2.z, q2.w, q3.x, q3.y, q3.z, q3.w};
        v16h ap, an;
#pragma unroll
        for (int j = 0; j < 16; ++j) {
            ap[j] = (_Float16)((vals[j] >= 1) ? 1.0f : 0.0f);
            an[j] = (_Float16)((vals[j] == 0) ? 1.0f : 0.0f);
        }
        accp = __builtin_amdgcn_wmma_f32_16x16x32_f16(false, ap, false, ones,
                                                      (short)0, accp, false, false);
        accn = __builtin_amdgcn_wmma_f32_16x16x32_f16(false, an, false, ones,
                                                      (short)0, accn, false, false);
    }
    // sum of all 16x16 D entries = 16 * total, layout-independent
    float lp = accp[0]+accp[1]+accp[2]+accp[3]+accp[4]+accp[5]+accp[6]+accp[7];
    float ln = accn[0]+accn[1]+accn[2]+accn[3]+accn[4]+accn[5]+accn[6]+accn[7];
#pragma unroll
    for (int off = 16; off > 0; off >>= 1) {
        lp += __shfl_xor(lp, off, 32);
        ln += __shfl_xor(ln, off, 32);
    }
    if (lane == 0) {
        stats[b * 8 + 0] = (int)(lp * (1.0f / 16.0f) + 0.5f);  // n_pos
        stats[b * 8 + 1] = (int)(ln * (1.0f / 16.0f) + 0.5f);  // n_neg
    }
}

// ---------------- Kernel 3: exact k-th threshold via bit bisection ---------
__global__ void __launch_bounds__(256)
select_kernel(const float* __restrict__ r,
              const int*   __restrict__ lab,
              int* __restrict__ stats) {
    const int b = blockIdx.x;
    __shared__ unsigned int  s_key[NPT] __attribute__((aligned(16)));  // 32.4 KB
    __shared__ unsigned char s_cls[NPT];                               //  8.1 KB
    __shared__ int s_cnt;

    // keys are the raw float bits: async bulk-copy the row straight into LDS
    const float4* rrow = (const float4*)(r + (size_t)b * NPT);  // NPT/4 = 2025
    for (int q = threadIdx.x; q < NPT / 4; q += 256) {
        async_load_b128(lds_offset(&s_key[q * 4]), &rrow[q]);
    }
    wait_asynccnt0();
    for (int t = threadIdx.x; t < NPT; t += 256) {
        const int l = lab[b * PTPAD + t];
        s_cls[t] = (l >= 1) ? 1 : ((l == 0) ? 0 : 2);
    }
    __syncthreads();

    const int n_pos = stats[b * 8 + 0];
    const int n_neg = stats[b * 8 + 1];
    const int k_pos = min(NPOS_CAP, n_pos);
    const int k_neg = min(KTOT - k_pos, n_neg);

    unsigned int taus[2];
    const int ks[2]   = {k_pos, k_neg};
    const int clss[2] = {1, 0};
    for (int s = 0; s < 2; ++s) {
        const int k = ks[s], cls = clss[s];
        if (k <= 0) { taus[s] = 0xffffffffu; continue; }
        unsigned int lo = 0u, hi = 0x3f800000u;   // bits of [0,1) are monotone
        while (hi - lo > 1u) {
            const unsigned int mid = lo + ((hi - lo) >> 1);
            if (threadIdx.x == 0) s_cnt = 0;
            __syncthreads();
            int local = 0;
            for (int t = threadIdx.x; t < NPT; t += 256)
                local += (s_cls[t] == (unsigned char)cls && s_key[t] >= mid) ? 1 : 0;
#pragma unroll
            for (int off = 16; off > 0; off >>= 1) local += __shfl_xor(local, off, 32);
            if ((threadIdx.x & 31) == 0) atomicAdd(&s_cnt, local);
            __syncthreads();
            const int c = s_cnt;
            __syncthreads();
            if (c >= k) lo = mid; else hi = mid;
        }
        taus[s] = lo;   // largest u with count(>= u) >= k  ==> exactly k selected
    }
    if (threadIdx.x == 0) {
        stats[b * 8 + 2] = k_pos;
        stats[b * 8 + 3] = k_neg;
        stats[b * 8 + 4] = (int)taus[0];
        stats[b * 8 + 5] = (int)taus[1];
    }
}

// ---------------- Kernel 4: stable compaction (== top_k(prio)) -------------
__global__ void __launch_bounds__(256)
compact_kernel(const float* __restrict__ r,
               const int*   __restrict__ lab,
               const int*   __restrict__ stats,
               int* __restrict__ inds) {
    const int b = blockIdx.x;
    __shared__ int s_scan[256];
    __shared__ int s_base_sel, s_base_uns;

    const unsigned int tau_p = (unsigned int)stats[b * 8 + 4];
    const unsigned int tau_n = (unsigned int)stats[b * 8 + 5];
    const int n_sel_tot = stats[b * 8 + 2] + stats[b * 8 + 3];
    if (threadIdx.x == 0) { s_base_sel = 0; s_base_uns = 0; }
    __syncthreads();

    for (int chunk = 0; chunk < NPT; chunk += 256) {
        const int t = chunk + (int)threadIdx.x;
        const int valid = (t < NPT) ? 1 : 0;
        int sel = 0;
        if (valid) {
            const int l = lab[b * PTPAD + t];
            const unsigned int key = __float_as_uint(r[b * NPT + t]);
            sel = ((l >= 1 && key >= tau_p) || (l == 0 && key >= tau_n)) ? 1 : 0;
        }
        s_scan[threadIdx.x] = sel;
        __syncthreads();
        for (int off = 1; off < 256; off <<= 1) {
            const int v = ((int)threadIdx.x >= off) ? s_scan[threadIdx.x - off] : 0;
            __syncthreads();
            s_scan[threadIdx.x] += v;
            __syncthreads();
        }
        const int inc      = s_scan[threadIdx.x];
        const int tot_sel  = s_scan[255];
        const int base_sel = s_base_sel;
        const int base_uns = s_base_uns;
        const int ex_sel   = inc - sel;
        const int vbefore  = min((int)threadIdx.x, NPT - chunk);
        const int ex_uns   = vbefore - ex_sel;
        if (valid) {
            const int slot = sel ? (base_sel + ex_sel)
                                 : (n_sel_tot + base_uns + ex_uns);
            if (slot >= 0 && slot < KTOT) inds[b * KTOT + slot] = t;
        }
        __syncthreads();
        if (threadIdx.x == 0) {
            const int tv = min(256, NPT - chunk);
            s_base_sel += tot_sel;
            s_base_uns += tv - tot_sel;
        }
        __syncthreads();
    }
}

// ---------------- Kernel 5: gather + BoxCoder.encode -----------------------
__global__ void __launch_bounds__(256)
gather_kernel(const float* __restrict__ proposals,
              const float* __restrict__ gt_boxes,
              const int*   __restrict__ lab,
              const int*   __restrict__ midx,
              const int*   __restrict__ inds,
              float* __restrict__ out) {
    const int idx = blockIdx.x * 256 + threadIdx.x;   // [0, NB*KTOT)
    if (idx >= NB * KTOT) return;
    const int b = idx / KTOT;
    const int t = inds[idx];

    float4 pb = (t < NP) ? ((const float4*)proposals)[b * NP + t]
                         : ((const float4*)gt_boxes)[b * NG + (t - NP)];
    const int l = lab[b * PTPAD + t];
    const int m = midx[b * NPT + t];
    float4 gb = ((const float4*)gt_boxes)[b * NG + m];

    float* s_props  = out;                          // [NB,512,4]
    float* s_labels = out + NB * KTOT * 4;          // [NB,512]
    float* s_regs   = out + NB * KTOT * 5;          // [NB,512,4]

    ((float4*)s_props)[idx] = pb;
    s_labels[idx] = (float)l;

    const float pw = pb.z - pb.x, ph = pb.w - pb.y;
    const float px = pb.x + 0.5f * pw, py = pb.y + 0.5f * ph;
    const float gw = gb.z - gb.x, gh = gb.w - gb.y;
    const float gx = gb.x + 0.5f * gw, gy = gb.y + 0.5f * gh;
    float4 tgt;
    tgt.x = 10.0f * (gx - px) / pw;
    tgt.y = 10.0f * (gy - py) / ph;
    tgt.z = 5.0f * logf(gw / pw);
    tgt.w = 5.0f * logf(gh / ph);
    ((float4*)s_regs)[idx] = tgt;
}

// ---------------------------------------------------------------------------
extern "C" void kernel_launch(void* const* d_in, const int* in_sizes, int n_in,
                              void* d_out, int out_size, void* d_ws, size_t ws_size,
                              hipStream_t stream) {
    (void)in_sizes; (void)n_in; (void)out_size; (void)ws_size;
    const float* proposals = (const float*)d_in[0];
    const float* gt_boxes  = (const float*)d_in[1];
    const int*   gt_labels = (const int*)d_in[2];

    char* ws = (char*)d_ws;
    const size_t SZR = (size_t)NB * NPT * sizeof(float);    // 1,036,800 B
    const size_t SZL = (size_t)NB * PTPAD * sizeof(int);    // 1,048,576 B
    float* r     = (float*)(ws);
    int*   lab   = (int*)  (ws + SZR);
    int*   midx  = (int*)  (ws + SZR + SZL);
    int*   stats = (int*)  (ws + 2 * SZR + SZL);            // NB*8 ints
    int*   inds  = (int*)  (ws + 2 * SZR + SZL + 4096);     // NB*KTOT ints

    match_kernel  <<<dim3(PTPAD / 256, NB), 256, 0, stream>>>(
        proposals, gt_boxes, gt_labels, r, lab, midx);
    count_kernel  <<<NB, 32, 0, stream>>>(lab, stats);
    select_kernel <<<NB, 256, 0, stream>>>(r, lab, stats);
    compact_kernel<<<NB, 256, 0, stream>>>(r, lab, stats, inds);
    gather_kernel <<<(NB * KTOT + 255) / 256, 256, 0, stream>>>(
        proposals, gt_boxes, lab, midx, inds, (float*)d_out);
}
